// SO2Linear_7232724927039
// MI455X (gfx1250) — matrix operand order repaired and checked by reading, not compiled, not run on hardware
//
#include <hip/hip_runtime.h>

// ---------------- problem constants ----------------
#define T_ENT 145     // sparse-product entries
#define DIM   25      // sum of (2l+1), l=0..4
#define CIN   128
#define COUT  128
#define NTILE 16      // rows of x per workgroup (WMMA M)
#define LSTR  136     // padded ushort stride per (row,n) line in LDS (bank-conflict free, 16B aligned frags)
#define SMEM_BYTES (2 * DIM * NTILE * LSTR * 2)   // hi + lo bf16 x-tile = 217,600 B (< 320 KB WGP LDS)

typedef __attribute__((ext_vector_type(16))) __bf16 v16bf;
typedef __attribute__((ext_vector_type(8)))  float  v8f;

union Frag {
  uint4 q[2];
  v16bf v;
};

// ---------------- compile-time sparse tables (sorted by output row) ----------------
struct Tab {
  int   in[T_ENT];
  int   out[T_ENT];
  int   w[T_ENT];
  float coef[T_ENT];
  int   rs[DIM + 1];   // row-start offsets into entry list
};

constexpr Tab make_tab() {
  Tab t{};
  constexpr int off[5] = {0, 1, 4, 9, 16};
  constexpr float pn = 0.44721359549995793f;  // 1/sqrt(5) path norm
  int w = 0, k = 0;
  for (int lo = 0; lo < 5; ++lo)
    for (int li = 0; li < 5; ++li) {
      int oi = off[li], oo = off[lo];
      t.in[k] = oi + li; t.out[k] = oo + lo; t.w[k] = w; t.coef[k] = pn; ++k; ++w;
      int mm = li < lo ? li : lo;
      for (int m = 1; m <= mm; ++m) {
        int ip = oi + li + m, im = oi + li - m;
        int op = oo + lo + m, om = oo + lo - m;
        int wre = w, wim = w + 1; w += 2;
        t.in[k] = ip; t.out[k] = op; t.w[k] = wre; t.coef[k] =  pn; ++k;
        t.in[k] = im; t.out[k] = om; t.w[k] = wre; t.coef[k] =  pn; ++k;
        t.in[k] = ip; t.out[k] = om; t.w[k] = wim; t.coef[k] =  pn; ++k;
        t.in[k] = im; t.out[k] = op; t.w[k] = wim; t.coef[k] = -pn; ++k;
      }
    }
  // stable insertion sort by output row -> one live accumulator at a time
  for (int i = 1; i < T_ENT; ++i) {
    int a = t.in[i], b = t.out[i], c = t.w[i]; float f = t.coef[i];
    int j = i - 1;
    while (j >= 0 && t.out[j] > b) {
      t.in[j + 1] = t.in[j]; t.out[j + 1] = t.out[j];
      t.w[j + 1]  = t.w[j];  t.coef[j + 1] = t.coef[j];
      --j;
    }
    t.in[j + 1] = a; t.out[j + 1] = b; t.w[j + 1] = c; t.coef[j + 1] = f;
  }
  int p = 0;
  for (int r = 0; r <= DIM; ++r) {
    while (p < T_ENT && t.out[p] < r) ++p;
    t.rs[r] = p;
  }
  return t;
}

__constant__ Tab d_tab = make_tab();

// ---------------- bf16 split helpers ----------------
__device__ __forceinline__ unsigned f2bf_bits(float f) {
  unsigned u = __float_as_uint(f);
  return (u + 0x7FFFu + ((u >> 16) & 1u)) >> 16;   // round-to-nearest-even
}
__device__ __forceinline__ float bf_to_f(unsigned hi16) {
  return __uint_as_float(hi16 << 16);
}

// ---------------- kernel 1: fold coef into weights, split hi/lo, pre-swizzle into B-fragment layout ----------------
// layout: [entry t][kb 0..3][db 0..7][lane 0..31][16 bf16]
//   element (lane l, slot j): K = kb*32 + (l>=16)*16 + j, N(col d) = db*16 + (l&15)
__global__ __launch_bounds__(256) void so2_prep_weights(const float* __restrict__ w,
                                                        unsigned short* __restrict__ whi,
                                                        unsigned short* __restrict__ wlo) {
  int idx = blockIdx.x * 256 + threadIdx.x;
  if (idx >= T_ENT * 16384) return;
  int j  = idx & 15;
  int l  = (idx >> 4) & 31;
  int db = (idx >> 9) & 7;
  int kb = (idx >> 12) & 3;
  int t  = idx >> 14;
  int c = kb * 32 + (l >> 4) * 16 + j;
  int d = db * 16 + (l & 15);
  float v = d_tab.coef[t] * w[(size_t)d_tab.w[t] * (CIN * COUT) + (size_t)c * COUT + d];
  unsigned hb = f2bf_bits(v);
  float lo = v - bf_to_f(hb);
  unsigned lb = f2bf_bits(lo);
  whi[idx] = (unsigned short)hb;
  wlo[idx] = (unsigned short)lb;
}

// ---------------- kernel 2: main SO(2)-linear via bf16x3 WMMA ----------------
// grid: N/16 workgroups, 256 threads (8 wave32); wave owns d-block db = wave_id (16 cols).
__global__ __launch_bounds__(256) void so2_main(const float* __restrict__ x,
                                                const unsigned short* __restrict__ whi,
                                                const unsigned short* __restrict__ wlo,
                                                float* __restrict__ out) {
  extern __shared__ unsigned short smem[];
  unsigned short* xh = smem;
  unsigned short* xl = smem + DIM * NTILE * LSTR;

  const int tid  = threadIdx.x;
  const int tile = blockIdx.x;
  const float* xt = x + (size_t)tile * NTILE * DIM * CIN;

  // ---- stage x-tile into LDS as bf16 hi/lo (packed 2 elems per uint store) ----
  for (int p = tid; p < DIM * NTILE * (CIN / 2); p += 256) {
    int c2 = p & 63;            // pair index along C
    int n  = (p >> 6) & 15;     // local row in n-tile
    int r  = p >> 10;           // irrep component
    float2 v = *reinterpret_cast<const float2*>(xt + ((size_t)n * DIM + r) * CIN + c2 * 2);
    unsigned h0 = f2bf_bits(v.x), h1 = f2bf_bits(v.y);
    unsigned l0 = f2bf_bits(v.x - bf_to_f(h0));
    unsigned l1 = f2bf_bits(v.y - bf_to_f(h1));
    int e = (r * NTILE + n) * LSTR + c2 * 2;
    *reinterpret_cast<unsigned*>(xh + e) = h0 | (h1 << 16);
    *reinterpret_cast<unsigned*>(xl + e) = l0 | (l1 << 16);
  }
  __syncthreads();

  const int lane = tid & 31;
  const int db   = tid >> 5;        // wave id = d-block
  const int lm   = lane & 15;
  const int lh   = lane >> 4;       // K-half selector (contiguous k = 16*lh + s, same order as B)

  for (int r = 0; r < DIM; ++r) {
    v8f acc = 0.0f;
    const int e0 = d_tab.rs[r], e1 = d_tab.rs[r + 1];
    for (int e = e0; e < e1; ++e) {
      const int in = d_tab.in[e];   // uniform across wave
      const int abase = (in * NTILE + lm) * LSTR + lh * 16;
      const unsigned short* bh_p = whi + ((size_t)e * 32 + db) * 512 + lane * 16;
      const unsigned short* bl_p = wlo + ((size_t)e * 32 + db) * 512 + lane * 16;
      if (e + 1 < e1) {             // pull next entry's fragments toward L2/L0
        __builtin_prefetch(whi + ((size_t)(e + 1) * 32 + db) * 512 + lane * 16, 0, 1);
        __builtin_prefetch(wlo + ((size_t)(e + 1) * 32 + db) * 512 + lane * 16, 0, 1);
      }
#pragma unroll
      for (int kb = 0; kb < 4; ++kb) {
        Frag ah, al, bh, bl;
        const unsigned short* ap = xh + abase + kb * 32;
        ah.q[0] = *reinterpret_cast<const uint4*>(ap);        // K = kb*32 + koff   .. +7
        ah.q[1] = *reinterpret_cast<const uint4*>(ap + 8);    // K = kb*32 + 8+koff .. +15
        const unsigned short* alp = xl + abase + kb * 32;
        al.q[0] = *reinterpret_cast<const uint4*>(alp);
        al.q[1] = *reinterpret_cast<const uint4*>(alp + 8);
        const unsigned short* bp = bh_p + kb * 4096;
        bh.q[0] = *reinterpret_cast<const uint4*>(bp);
        bh.q[1] = *reinterpret_cast<const uint4*>(bp + 8);
        const unsigned short* blp = bl_p + kb * 4096;
        bl.q[0] = *reinterpret_cast<const uint4*>(blp);
        bl.q[1] = *reinterpret_cast<const uint4*>(blp + 8);

        acc = __builtin_amdgcn_wmma_f32_16x16x32_bf16(false, ah.v, false, bh.v,
                                                      (short)0, acc, false, false);
        acc = __builtin_amdgcn_wmma_f32_16x16x32_bf16(false, al.v, false, bh.v,
                                                      (short)0, acc, false, false);
        acc = __builtin_amdgcn_wmma_f32_16x16x32_bf16(false, ah.v, false, bl.v,
                                                      (short)0, acc, false, false);
      }
    }
    // C/D layout: VGPR v, lanes 0-15 -> M=v, lanes 16-31 -> M=8+v; col = db*16 + lm
    float* op = out + (((size_t)(tile * NTILE + lh * 8)) * DIM + r) * COUT + db * 16 + lm;
#pragma unroll
    for (int v = 0; v < 8; ++v) op[(size_t)v * DIM * COUT] = acc[v];
  }
}

// ---------------- launcher ----------------
extern "C" void kernel_launch(void* const* d_in, const int* in_sizes, int n_in,
                              void* d_out, int out_size, void* d_ws, size_t ws_size,
                              hipStream_t stream) {
  const float* x = (const float*)d_in[0];
  const float* w = (const float*)d_in[1];
  float* out = (float*)d_out;

  unsigned short* whi = (unsigned short*)d_ws;
  unsigned short* wlo = whi + (size_t)T_ENT * 16384;   // 2x 4.75 MB in workspace

  const int n      = in_sizes[0] / (DIM * CIN);        // 8192
  const int ntiles = n / NTILE;                        // 512

  (void)hipFuncSetAttribute(reinterpret_cast<const void*>(so2_main),
                            hipFuncAttributeMaxDynamicSharedMemorySize, SMEM_BYTES);

  so2_prep_weights<<<(T_ENT * 16384 + 255) / 256, 256, 0, stream>>>(w, whi, wlo);
  so2_main<<<ntiles, 256, SMEM_BYTES, stream>>>(x, whi, wlo, out);
}
